// UniversalKANLinear_11871289606591
// MI455X (gfx1250) — compile-verified, matrix operand order checked by reading, not run on hardware
//
#include <hip/hip_runtime.h>
#include <hip/hip_bf16.h>

// ---------------------------------------------------------------------------
// UniversalKAN linear (lshifted softplus) for MI455X / gfx1250.
//   out = (softplus([x,1]) - ln2) @ W^T     (8192 x 2048) x (2048 x 2049)
// f16-convert phi and W once, then f16 WMMA GEMM (f32 acc) with a 3-stage
// async-to-LDS ring (depth-2 prefetch, s_wait_asynccnt<=4), last two
// K-iterations peeled so the steady-state loop is branch-minimal.
// Workspace layout (requires ~42 MB):
//   [0)                phiH : 8192*2048 f16
//   [33,554,432)       Wh   : 2048*2048 f16
//   [41,943,040)       bias : 2048 f32   (= phi(1) * W[:,2048])
// ---------------------------------------------------------------------------

typedef __attribute__((ext_vector_type(16))) _Float16 v16h;
typedef __attribute__((ext_vector_type(8)))  _Float16 v8h;
typedef __attribute__((ext_vector_type(4)))  _Float16 v4h;
typedef __attribute__((ext_vector_type(8)))  float    v8f;

#define KAN_B   8192
#define KAN_K   2048
#define KAN_N   2048
#define LN2F    0.69314718055994530942f
#define PHI1F   0.62011450695827913f   /* ln(1+e) - ln2 = softplus(1)-ln2 */

static __device__ __forceinline__ float softplus_m_ln2(float x) {
    float t = __expf(-fabsf(x));
    return fmaxf(x, 0.0f) + __logf(1.0f + t) - LN2F;
}

// ---------------- pass 1a: phi = softplus(x) - ln2, f32 -> f16 -------------
__global__ __launch_bounds__(256) void phi_kernel(const float* __restrict__ x,
                                                  _Float16* __restrict__ phiH) {
    int t = blockIdx.x * blockDim.x + threadIdx.x;   // one float4 per thread
    float4 v = ((const float4*)x)[t];
    v4h o;
    o[0] = (_Float16)softplus_m_ln2(v.x);
    o[1] = (_Float16)softplus_m_ln2(v.y);
    o[2] = (_Float16)softplus_m_ln2(v.z);
    o[3] = (_Float16)softplus_m_ln2(v.w);
    ((v4h*)phiH)[t] = o;
}

// ---------------- pass 1b: W f32 -> f16, split bias column -----------------
__global__ __launch_bounds__(256) void wprep_kernel(const float* __restrict__ W,
                                                    _Float16* __restrict__ Wh,
                                                    float* __restrict__ bias) {
    int t = blockIdx.x * blockDim.x + threadIdx.x;   // over 2048*2049 elements
    int o = t / (KAN_K + 1);
    int i = t - o * (KAN_K + 1);
    float w = W[t];
    if (i < KAN_K) Wh[(size_t)o * KAN_K + i] = (_Float16)w;
    else           bias[o] = PHI1F * w;
}

// ---------------- pass 2: C = phiH @ Wh^T + bias ---------------------------
#define BM 128
#define BN 128
#define BK 32
#define LDT 40                       /* padded LDS row stride: 80 B        */
#define A_BYTES (BM * LDT * 2)       /* 10,240 B per A tile                */
#define STAGE_BYTES (2 * A_BYTES)    /* A + B per stage = 20,480 B         */
#define KTILES (KAN_K / BK)          /* 64 */

__global__ __launch_bounds__(256) void kan_gemm(const _Float16* __restrict__ A,
                                                const _Float16* __restrict__ Bm,
                                                const float* __restrict__ bias,
                                                float* __restrict__ C) {
    __shared__ __attribute__((aligned(16))) char smem[3 * STAGE_BYTES]; // 60 KB

    const int tid   = threadIdx.x;
    const int lane  = tid & 31;
    const int wave  = tid >> 5;            // 0..7
    const int waveM = wave >> 2;           // 0..1  -> 64-row slice
    const int waveN = wave & 3;            // 0..3  -> 32-col slice
    const int bm    = blockIdx.x * BM;
    const int bn    = blockIdx.y * BN;

    // --- async copy mapping: 256 threads, 4x16B chunks per 32-half row ---
    const int chunk = tid & 3;             // 8-half chunk within a row
    const int row0  = tid >> 2;            // 0..63 (also +64)

    const unsigned smem0  = (unsigned)(uintptr_t)&smem[0];
    const _Float16* gA0   = A  + (size_t)(bm + row0) * KAN_K + chunk * 8;
    const _Float16* gB0   = Bm + (size_t)(bn + row0) * KAN_K + chunk * 8;
    const unsigned  ldsRow = (unsigned)(row0 * LDT + chunk * 8) * 2u;

#define ISSUE_STAGE(kt, offBytes)                                              \
    do {                                                                       \
        const _Float16* ga = gA0 + (kt) * BK;                                  \
        const _Float16* gb = gB0 + (kt) * BK;                                  \
        unsigned la = smem0 + (offBytes) + ldsRow;                             \
        unsigned lb = la + A_BYTES;                                            \
        asm volatile("global_load_async_to_lds_b128 %0, %1, off"               \
                     :: "v"(la), "v"((unsigned long long)(uintptr_t)ga)        \
                     : "memory");                                              \
        asm volatile("global_load_async_to_lds_b128 %0, %1, off"               \
                     :: "v"((unsigned)(la + 64u * LDT * 2u)),                  \
                        "v"((unsigned long long)(uintptr_t)(ga + (size_t)64 * KAN_K)) \
                     : "memory");                                              \
        asm volatile("global_load_async_to_lds_b128 %0, %1, off"               \
                     :: "v"(lb), "v"((unsigned long long)(uintptr_t)gb)        \
                     : "memory");                                              \
        asm volatile("global_load_async_to_lds_b128 %0, %1, off"               \
                     :: "v"((unsigned)(lb + 64u * LDT * 2u)),                  \
                        "v"((unsigned long long)(uintptr_t)(gb + (size_t)64 * KAN_K)) \
                     : "memory");                                              \
    } while (0)

    union FragH { v16h v; v8h h[2]; };
    v8f acc[4][2] = {};

    const int half = lane >> 4;            // K half-select per ISA layout
    const int r    = lane & 15;            // M (or N) within 16-wide tile

    // per-thread fragment byte offsets within a stage (loop-invariant)
    unsigned aOff[4], bOff[2];
#pragma unroll
    for (int tm = 0; tm < 4; ++tm)
        aOff[tm] = (unsigned)(((waveM * 64 + tm * 16 + r) * LDT + half * 8) * 2);
#pragma unroll
    for (int tn = 0; tn < 2; ++tn)
        bOff[tn] = (unsigned)(A_BYTES + ((waveN * 32 + tn * 16 + r) * LDT + half * 8) * 2);

    // one K-stage of fragment loads + 8 WMMAs from the buffer at oCur
    auto compute_stage = [&](unsigned oCur) {
        FragH fa[4], fb[2];
#pragma unroll
        for (int tm = 0; tm < 4; ++tm) {
            const char* p = smem + oCur + aOff[tm];
            fa[tm].h[0] = *(const v8h*)p;          // K = half*8 .. +7
            fa[tm].h[1] = *(const v8h*)(p + 32);   // K = 16+half*8 .. +7
        }
#pragma unroll
        for (int tn = 0; tn < 2; ++tn) {
            const char* p = smem + oCur + bOff[tn];
            fb[tn].h[0] = *(const v8h*)p;
            fb[tn].h[1] = *(const v8h*)(p + 32);
        }
#pragma unroll
        for (int tm = 0; tm < 4; ++tm)
#pragma unroll
            for (int tn = 0; tn < 2; ++tn)
                acc[tm][tn] = __builtin_amdgcn_wmma_f32_16x16x32_f16(
                    false, fa[tm].v, false, fb[tn].v,
                    (short)0, acc[tm][tn], false, false);
    };

    // 3-stage ring, depth-2 prefetch; last two iterations peeled.
    unsigned oC = 0, oN = STAGE_BYTES, oF = 2 * STAGE_BYTES;
    ISSUE_STAGE(0, oC);
    ISSUE_STAGE(1, oN);

#pragma unroll 1
    for (int kt = 0; kt < KTILES - 2; ++kt) {
        // async loads retire in order per wave: <=4 outstanding means the
        // oldest stage (kt) has fully landed; the kt+1 stage may still fly.
        asm volatile("s_wait_asynccnt 0x4" ::: "memory");
        __syncthreads();   // stage kt resident everywhere; buffer oF free
        ISSUE_STAGE(kt + 2, oF);
        compute_stage(oC);
        unsigned t = oC; oC = oN; oN = oF; oF = t;   // rotate ring
    }

    // kt = KTILES-2 : stage KTILES-1 still in flight, nothing new to issue
    asm volatile("s_wait_asynccnt 0x4" ::: "memory");
    __syncthreads();
    compute_stage(oC);
    { unsigned t = oC; oC = oN; oN = oF; oF = t; }

    // kt = KTILES-1 : drain everything
    asm volatile("s_wait_asynccnt 0x0" ::: "memory");
    __syncthreads();
    compute_stage(oC);

    // ---------------- epilogue: C = acc + bias[col] ------------------------
#pragma unroll
    for (int tm = 0; tm < 4; ++tm) {
#pragma unroll
        for (int tn = 0; tn < 2; ++tn) {
            int rowb = bm + waveM * 64 + tm * 16 + half * 8;   // M = rowb + j
            int col  = bn + waveN * 32 + tn * 16 + r;          // N = lane&15
            float bv = bias[col];
#pragma unroll
            for (int j = 0; j < 8; ++j)
                C[(size_t)(rowb + j) * KAN_N + col] = acc[tm][tn][j] + bv;
        }
    }
#undef ISSUE_STAGE
}

// ---------------------------------------------------------------------------
extern "C" void kernel_launch(void* const* d_in, const int* in_sizes, int n_in,
                              void* d_out, int out_size, void* d_ws, size_t ws_size,
                              hipStream_t stream) {
    const float* x = (const float*)d_in[0];          // 8192 x 2048 f32
    const float* W = (const float*)d_in[1];          // 2048 x 2049 f32
    float* out = (float*)d_out;                      // 8192 x 2048 f32

    _Float16* phiH = (_Float16*)d_ws;
    _Float16* Wh   = (_Float16*)((char*)d_ws + (size_t)KAN_B * KAN_K * 2);
    float*    bias = (float*)((char*)d_ws + (size_t)KAN_B * KAN_K * 2
                                          + (size_t)KAN_N * KAN_K * 2);

    phi_kernel<<<(KAN_B * KAN_K / 4) / 256, 256, 0, stream>>>(x, phiH);
    wprep_kernel<<<(KAN_N * (KAN_K + 1)) / 256, 256, 0, stream>>>(W, Wh, bias);

    dim3 grid(KAN_B / BM, KAN_N / BN);               // 64 x 16
    kan_gemm<<<grid, 256, 0, stream>>>(phiH, Wh, bias, out);
}
